// HL_filter_87247965651030
// MI455X (gfx1250) — compile-verified, matrix operand order checked by reading, not run on hardware
//
#include <hip/hip_runtime.h>
#include <hip/hip_bf16.h>

#define NT_N 100000
#define NS_N 400000
#define ET_N 1600000
#define ES_N 3200000
#define P_N  800000
#define DIN  64
#define FEAT 32
#define SLOPE 0.1f
#define EPSV 1e-5f

typedef __attribute__((ext_vector_type(16))) _Float16 v16h;
typedef __attribute__((ext_vector_type(8)))  float    v8f;

// ---------------------------------------------------------------------------
// WMMA helpers (CDNA5 16x16x32 f16 -> f32). ISA 7.12.2 16-bit A layout:
// lane L (0-15): row M=L, K in {0..7,16..23}; lane L+16: same row, K in
// {8..15,24..31}.  Element e = 2v+p:  K = 2v+p + (v>=4 ? 8:0) + (hi ? 8:0).
// B mirrors A with N = lane%16.
// ---------------------------------------------------------------------------
__device__ __forceinline__ int kidx(int e, int hi) {
    int v = e >> 1, p = e & 1;
    return 2 * v + p + ((v >= 4) ? 8 : 0) + (hi ? 8 : 0);
}

__device__ __forceinline__ v8f wmma_f16(v16h a, v16h b, v8f c) {
    return __builtin_amdgcn_wmma_f32_16x16x32_f16(false, a, false, b,
                                                  (short)0, c, false, false);
}

// A fragment from row-major f32 [.. x ld], rows r..r+15, cols k0..k0+31
__device__ __forceinline__ v16h load_a_g(const float* src, int ld, int r,
                                         int k0, float sc, int hi) {
    const float* p = src + (size_t)r * ld + k0;
    v16h a;
#pragma unroll
    for (int e = 0; e < 16; ++e) a[e] = (_Float16)(p[kidx(e, hi)] * sc);
    return a;
}

// B fragments (cols 0..15 and 16..31) from f32 weight tile W[k][32] in LDS
__device__ __forceinline__ void load_b2(const float* W, int k0, int rl, int hi,
                                        v16h& b0, v16h& b1) {
#pragma unroll
    for (int e = 0; e < 16; ++e) {
        int k = k0 + kidx(e, hi);
        b0[e] = (_Float16)W[k * 32 + rl];
        b1[e] = (_Float16)W[k * 32 + 16 + rl];
    }
}

// D store (f32, ld=32) fused with per-column sum/sumsq accumulation.
// C/D layout: lane L: N=L%16; VGPR i: M = i + (L>=16 ? 8 : 0).
__device__ __forceinline__ void store_d_stats(v8f c, float* Y, int row0,
                                              int hi, int n, float* colsum,
                                              float* colsq) {
    float s = 0.f, ss = 0.f;
#pragma unroll
    for (int i = 0; i < 8; ++i) {
        float v = c[i];
        Y[(size_t)(row0 + 8 * hi + i) * 32 + n] = v;
        s += v;
        ss += v * v;
    }
    s  += __shfl_xor(s, 16, 32);   // pair lanes L <-> L+16 (same column)
    ss += __shfl_xor(ss, 16, 32);
    if (hi == 0) {
        atomicAdd(&colsum[n], s);
        atomicAdd(&colsq[n], ss);
    }
}

// ---------------------------------------------------------------------------
// Utility kernels
// ---------------------------------------------------------------------------
__global__ void k_zero(float* p, long n) {
    long i = (long)blockIdx.x * blockDim.x + threadIdx.x;
    long st = (long)gridDim.x * blockDim.x;
    for (; i < n; i += st) p[i] = 0.f;
}

// copy [n x 64] into xcat (row stride 128), cols 0..63
__global__ void k_init_xcat(const float* __restrict__ x0,
                            float* __restrict__ xcat, long n64) {
    long i = (long)blockIdx.x * blockDim.x + threadIdx.x;
    if (i < n64) {
        long r = i >> 6;
        int  c = (int)(i & 63);
        xcat[r * 128 + c] = x0[i];
    }
}

// Laguerre closed form: Wa = W0+W1+W2+W3 ; Wbneg = -(W1 + 2W2 + 3W3)
__global__ void k_combine(const float* W0, const float* W1, const float* W2,
                          const float* W3, float* Wa, float* Wbneg) {
    int i = blockIdx.x * blockDim.x + threadIdx.x;
    if (i < 1024) {
        Wa[i]    = W0[i] + W1[i] + W2[i] + W3[i];
        Wbneg[i] = -(W1[i] + 2.f * W2[i] + 3.f * W3[i]);
    }
}

// msg[dst[p]] += |w[p]| * xsrc[src[p]]   (xsrc row stride 128, msg stride d)
__global__ void k_par_scatter(const int* __restrict__ dst,
                              const int* __restrict__ src,
                              const float* __restrict__ w,
                              const float* __restrict__ xsrc,
                              float* msg, int np, int d) {
    int i  = blockIdx.x * blockDim.x + threadIdx.x;
    int st = gridDim.x * blockDim.x;
    for (; i < np; i += st) {
        float aw = fabsf(w[i]);
        const float* xr = xsrc + (size_t)src[i] * 128;
        float*       mr = msg + (size_t)dst[i] * d;
        __builtin_prefetch(xr, 0, 0);
#pragma unroll 8
        for (int f = 0; f < d; ++f) atomicAdd(&mr[f], aw * xr[f]);
    }
}

// Ax[col[e]] += w[e] * x[row[e]]   (32 features, stride 32)
__global__ void k_edge_scatter(const int* __restrict__ row,
                               const int* __restrict__ col,
                               const float* __restrict__ w,
                               const float* __restrict__ x,
                               float* ax, int ne) {
    int i  = blockIdx.x * blockDim.x + threadIdx.x;
    int st = gridDim.x * blockDim.x;
    for (; i < ne; i += st) {
        float wv = w[i];
        const float* xr = x + (size_t)row[i] * 32;
        float*       ar = ax + (size_t)col[i] * 32;
        __builtin_prefetch(xr, 0, 0);
#pragma unroll
        for (int f = 0; f < 32; ++f) atomicAdd(&ar[f], wv * xr[f]);
    }
}

// stats layout: [0:32) sum, [32:64) sumsq, [64:96) scale, [96:128) shift
__global__ void k_finalize(float* stats, float invN, const float* __restrict__ g,
                           const float* __restrict__ be) {
    int c = threadIdx.x;
    if (c < 32) {
        float m   = stats[c] * invN;
        float var = stats[32 + c] * invN - m * m;
        float sc  = g[c] * rsqrtf(var + EPSV);
        stats[64 + c] = sc;
        stats[96 + c] = be[c] - m * sc;
        stats[c] = 0.f;           // re-arm accumulators for next stage
        stats[32 + c] = 0.f;
    }
}

// y = act(Y*scale + shift) -> dst[r*ldd + coff + c]  (slope=0 -> ReLU)
__global__ void k_bn_act(const float* Y, const float* __restrict__ stats,
                         float* dst, int ldd, int coff, float slope, long n32) {
    long i = (long)blockIdx.x * blockDim.x + threadIdx.x;
    if (i < n32) {
        int  c = (int)(i & 31);
        long r = i >> 5;
        float v = fmaf(Y[i], stats[64 + c], stats[96 + c]);
        v = v > 0.f ? v : slope * v;
        dst[r * ldd + coff + c] = v;
    }
}

// ---------------------------------------------------------------------------
// Layer 1: Y = [msg*rowscale , xcat] @ W1  ([N x 2d] @ [2d x 32]) + stats
// ---------------------------------------------------------------------------
__global__ void __launch_bounds__(256) k_mlp_l1(
    const float* __restrict__ msg, const float* __restrict__ Dvec, float cscale,
    const float* __restrict__ xcat, const float* __restrict__ W1,
    float* __restrict__ Y, float* stats, int N, int d) {
    __shared__ float lw[192 * 32];                  // up to 2d=192 rows of W1
    int tid = threadIdx.x;
    int nk  = 2 * d;
    for (int i = tid; i < nk * 32; i += 256) lw[i] = W1[i];
    __syncthreads();

    int wave = tid >> 5, lane = tid & 31;
    int row0 = blockIdx.x * 128 + wave * 16;
    if (row0 >= N) return;                          // wave-uniform, EXEC stays full
    int hi = lane >> 4, rl = lane & 15;
    int r  = row0 + rl;
    float sc = Dvec ? (1.0f / Dvec[r]) : cscale;    // scale applies to msg half only

    v8f acc0 = {}, acc1 = {};
    for (int k0 = 0; k0 < d; k0 += 32) {            // msg part, W1 rows [0,d)
        v16h a = load_a_g(msg, d, r, k0, sc, hi);
        v16h b0, b1;
        load_b2(lw, k0, rl, hi, b0, b1);
        acc0 = wmma_f16(a, b0, acc0);
        acc1 = wmma_f16(a, b1, acc1);
    }
    for (int k0 = 0; k0 < d; k0 += 32) {            // xcat part, W1 rows [d,2d)
        v16h a = load_a_g(xcat, 128, r, k0, 1.0f, hi);
        v16h b0, b1;
        load_b2(lw, d + k0, rl, hi, b0, b1);
        acc0 = wmma_f16(a, b0, acc0);
        acc1 = wmma_f16(a, b1, acc1);
    }
    store_d_stats(acc0, Y, row0, hi, rl, stats, stats + 32);
    store_d_stats(acc1, Y, row0, hi, 16 + rl, stats, stats + 32);
}

// ---------------------------------------------------------------------------
// Layer 2 (in place): H <- relu(bn1(H)) @ W2   + stats of the new values
// ---------------------------------------------------------------------------
__global__ void __launch_bounds__(256) k_mlp_l2(float* H,
                                                const float* __restrict__ W2,
                                                float* stats, int N) {
    __shared__ float lw[1024];
    __shared__ float lsc[32], lsh[32];
    int tid = threadIdx.x;
    for (int i = tid; i < 1024; i += 256) lw[i] = W2[i];
    if (tid < 32) { lsc[tid] = stats[64 + tid]; lsh[tid] = stats[96 + tid]; }
    __syncthreads();

    int wave = tid >> 5, lane = tid & 31;
    int row0 = blockIdx.x * 128 + wave * 16;
    if (row0 >= N) return;
    int hi = lane >> 4, rl = lane & 15;
    const float* src = H + (size_t)(row0 + rl) * 32;

    v16h a;
#pragma unroll
    for (int e = 0; e < 16; ++e) {
        int k   = kidx(e, hi);
        float v = fmaf(src[k], lsc[k], lsh[k]);     // BN affine
        a[e]    = (_Float16)(v > 0.f ? v : 0.f);    // ReLU
    }
    v16h b0, b1;
    load_b2(lw, 0, rl, hi, b0, b1);
    v8f acc0 = {}, acc1 = {};
    acc0 = wmma_f16(a, b0, acc0);
    acc1 = wmma_f16(a, b1, acc1);
    store_d_stats(acc0, H, row0, hi, rl, stats, stats + 32);
    store_d_stats(acc1, H, row0, hi, 16 + rl, stats, stats + 32);
}

// ---------------------------------------------------------------------------
// HL conv closed form: Y = X @ Wa + AX @ Wbneg   + stats
// ---------------------------------------------------------------------------
__global__ void __launch_bounds__(256) k_conv(const float* __restrict__ X,
                                              const float* __restrict__ AX,
                                              const float* __restrict__ Wa,
                                              const float* __restrict__ Wb,
                                              float* __restrict__ Y,
                                              float* stats, int N) {
    __shared__ float lwa[1024], lwb[1024];
    int tid = threadIdx.x;
    for (int i = tid; i < 1024; i += 256) { lwa[i] = Wa[i]; lwb[i] = Wb[i]; }
    __syncthreads();

    int wave = tid >> 5, lane = tid & 31;
    int row0 = blockIdx.x * 128 + wave * 16;
    if (row0 >= N) return;
    int hi = lane >> 4, rl = lane & 15;
    int r  = row0 + rl;

    v8f acc0 = {}, acc1 = {};
    {
        v16h a = load_a_g(X, 32, r, 0, 1.0f, hi);
        v16h b0, b1;
        load_b2(lwa, 0, rl, hi, b0, b1);
        acc0 = wmma_f16(a, b0, acc0);
        acc1 = wmma_f16(a, b1, acc1);
    }
    {
        v16h a = load_a_g(AX, 32, r, 0, 1.0f, hi);
        v16h b0, b1;
        load_b2(lwb, 0, rl, hi, b0, b1);
        acc0 = wmma_f16(a, b0, acc0);
        acc1 = wmma_f16(a, b1, acc1);
    }
    store_d_stats(acc0, Y, row0, hi, rl, stats, stats + 32);
    store_d_stats(acc1, Y, row0, hi, 16 + rl, stats, stats + 32);
}

// ---------------------------------------------------------------------------
static inline int ceil_div(int a, int b) { return (a + b - 1) / b; }

extern "C" void kernel_launch(void* const* d_in, const int* in_sizes, int n_in,
                              void* d_out, int out_size, void* d_ws,
                              size_t ws_size, hipStream_t stream) {
    if (n_in < 10) return;
    const float* x_t0 = (const float*)d_in[0];
    const int*   eit  = (const int*)d_in[1];
    const float* wt   = (const float*)d_in[2];
    const float* x_s0 = (const float*)d_in[3];
    const int*   eis  = (const int*)d_in[4];
    const float* wse  = (const float*)d_in[5];
    const int*   pidx = (const int*)d_in[6];
    const float* pw   = (const float*)d_in[7];
    const float* Dv   = (const float*)d_in[8];

    // params pytree flattened (jax sorts dict keys):
    // per channel j (30 leaves, base 9+30j):
    //  0..3 Ws_s[0..3], 4..7 Ws_t[0..3], 8 b_s, 9 b_t, 10 be_s, 11 be_t,
    //  12 g_s, 13 g_t, 14..21 msi_edge{W1,W2,b1,b2,be1,be2,g1,g2},
    //  22..29 msi_node{W1,W2,b1,b2,be1,be2,g1,g2}
    auto prm = [&](int j, int off) -> const float* {
        int idx = 9 + 30 * j + off;
        return (idx < n_in) ? (const float*)d_in[idx] : nullptr;
    };

    float* xcat_t = (float*)d_out;                    // [N_T x 128]
    float* xcat_s = xcat_t + (size_t)NT_N * 128;      // [N_S x 128]

    float* ws    = (float*)d_ws;
    float* msg   = ws;                                // N_S * 96
    float* hbuf  = msg + (size_t)NS_N * 96;           // N_S * 32
    float* axbuf = hbuf + (size_t)NS_N * 32;          // N_S * 32
    float* wcomb = axbuf + (size_t)NS_N * 32;         // 8 * 1024 (Wa/Wbneg x4)
    float* stats = wcomb + 8 * 1024;                  // 128
    float* ybuf  = msg;                               // conv out aliases msg

    k_zero<<<1, 128, 0, stream>>>(stats, 128);
    k_init_xcat<<<ceil_div(NT_N * 64, 256), 256, 0, stream>>>(x_t0, xcat_t,
                                                              (long)NT_N * 64);
    k_init_xcat<<<ceil_div(NS_N * 64, 256), 256, 0, stream>>>(x_s0, xcat_s,
                                                              (long)NS_N * 64);
    for (int j = 0; j < 2; ++j) {                     // combined conv weights
        k_combine<<<4, 256, 0, stream>>>(prm(j, 4), prm(j, 5), prm(j, 6),
                                         prm(j, 7), wcomb + j * 2048,
                                         wcomb + j * 2048 + 1024);          // t
        k_combine<<<4, 256, 0, stream>>>(prm(j, 0), prm(j, 1), prm(j, 2),
                                         prm(j, 3), wcomb + (2 + j) * 2048,
                                         wcomb + (2 + j) * 2048 + 1024);    // s
    }

    for (int j = 0; j < 2; ++j) {
        int d = DIN + FEAT * j;

        // ================= T pipeline =================
        k_zero<<<2048, 256, 0, stream>>>(msg, (long)NT_N * d);
        k_par_scatter<<<2048, 256, 0, stream>>>(pidx, pidx + P_N, pw, xcat_s,
                                                msg, P_N, d);
        k_mlp_l1<<<ceil_div(NT_N, 128), 256, 0, stream>>>(
            msg, Dv, 0.f, xcat_t, prm(j, 22), hbuf, stats, NT_N, d);
        k_finalize<<<1, 32, 0, stream>>>(stats, 1.0f / NT_N, prm(j, 28),
                                         prm(j, 26));
        k_mlp_l2<<<ceil_div(NT_N, 128), 256, 0, stream>>>(hbuf, prm(j, 23),
                                                          stats, NT_N);
        k_finalize<<<1, 32, 0, stream>>>(stats, 1.0f / NT_N, prm(j, 29),
                                         prm(j, 27));
        k_bn_act<<<ceil_div(NT_N * 32, 256), 256, 0, stream>>>(
            hbuf, stats, hbuf, 32, 0, 0.f, (long)NT_N * 32);
        k_zero<<<2048, 256, 0, stream>>>(axbuf, (long)NT_N * 32);
        k_edge_scatter<<<2048, 256, 0, stream>>>(eit, eit + ET_N, wt, hbuf,
                                                 axbuf, ET_N);
        k_conv<<<ceil_div(NT_N, 128), 256, 0, stream>>>(
            hbuf, axbuf, wcomb + j * 2048, wcomb + j * 2048 + 1024, ybuf,
            stats, NT_N);
        k_finalize<<<1, 32, 0, stream>>>(stats, 1.0f / NT_N, prm(j, 13),
                                         prm(j, 11));
        k_bn_act<<<ceil_div(NT_N * 32, 256), 256, 0, stream>>>(
            ybuf, stats, xcat_t, 128, 64 + 32 * j, SLOPE, (long)NT_N * 32);

        // ================= S pipeline =================
        k_zero<<<4096, 256, 0, stream>>>(msg, (long)NS_N * d);
        k_par_scatter<<<2048, 256, 0, stream>>>(pidx + P_N, pidx, pw, xcat_t,
                                                msg, P_N, d);
        k_mlp_l1<<<ceil_div(NS_N, 128), 256, 0, stream>>>(
            msg, nullptr, 0.5f, xcat_s, prm(j, 14), hbuf, stats, NS_N, d);
        k_finalize<<<1, 32, 0, stream>>>(stats, 1.0f / NS_N, prm(j, 20),
                                         prm(j, 18));
        k_mlp_l2<<<ceil_div(NS_N, 128), 256, 0, stream>>>(hbuf, prm(j, 15),
                                                          stats, NS_N);
        k_finalize<<<1, 32, 0, stream>>>(stats, 1.0f / NS_N, prm(j, 21),
                                         prm(j, 19));
        k_bn_act<<<ceil_div(NS_N * 32, 256), 256, 0, stream>>>(
            hbuf, stats, hbuf, 32, 0, 0.f, (long)NS_N * 32);
        k_zero<<<4096, 256, 0, stream>>>(axbuf, (long)NS_N * 32);
        k_edge_scatter<<<4096, 256, 0, stream>>>(eis, eis + ES_N, wse, hbuf,
                                                 axbuf, ES_N);
        k_conv<<<ceil_div(NS_N, 128), 256, 0, stream>>>(
            hbuf, axbuf, wcomb + (2 + j) * 2048, wcomb + (2 + j) * 2048 + 1024,
            ybuf, stats, NS_N);
        k_finalize<<<1, 32, 0, stream>>>(stats, 1.0f / NS_N, prm(j, 12),
                                         prm(j, 10));
        k_bn_act<<<ceil_div(NS_N * 32, 256), 256, 0, stream>>>(
            ybuf, stats, xcat_s, 128, 64 + 32 * j, SLOPE, (long)NS_N * 32);
    }
}
